// MultiHeadAttentionBlock_8959301779755
// MI455X (gfx1250) — matrix operand order repaired and checked by reading, not compiled, not run on hardware
//
#include <hip/hip_runtime.h>
#include <hip/hip_bf16.h>

typedef __attribute__((ext_vector_type(16))) __bf16 v16bf;
typedef __attribute__((ext_vector_type(8)))  __bf16 v8bf;
typedef __attribute__((ext_vector_type(8)))  float  v8f;

// ---- CDNA5 async global->LDS copy (ASYNCcnt path), via inline asm ----
// global_load_async_to_lds_b128: VDST = per-lane LDS byte address (32-bit),
// VADDR = 64-bit global address (see cdna5_isa 15.18.3, opcode 98).
// Flat shared-memory pointers truncate to valid LDS addresses (addr[31:0]).
__device__ __forceinline__ void async_copy_b128(const void* g, void* l) {
    unsigned lds = (unsigned)(uintptr_t)l;
    asm volatile("global_load_async_to_lds_b128 %0, %1, off"
                 :: "v"(lds), "v"(g) : "memory");
}

__device__ __forceinline__ void wait_async0() {
    asm volatile("s_wait_asynccnt 0x0" ::: "memory");
}

constexpr int B_ = 2, S_ = 2048, D_ = 1024, H_ = 16, DK_ = 64;
constexpr int M_ = B_ * S_;           // 4096 rows

// ---------------- GEMM: C[M,D] = A[M,D] @ W[D,D] + bias ----------------
constexpr int BM = 128, BN = 64, BK = 32;
constexpr int LDA = 40;   // bf16 elems; 80B row stride (16B aligned)
constexpr int LDB = 48;   // bf16 elems; 96B row stride (32B aligned)

template<bool A_BF16, bool SPLIT_HEADS>
__global__ __launch_bounds__(256)
void gemm_bias_kernel(const void* __restrict__ Ap,
                      const float* __restrict__ W,
                      const float* __restrict__ bias,
                      void* __restrict__ outp)
{
    __shared__ __bf16 As[BM * LDA];
    __shared__ __bf16 Bs[BN * LDB];   // stored transposed: Bs[n][k]

    const int tid  = threadIdx.x;
    const int n0   = blockIdx.x * BN;
    const int m0   = blockIdx.y * BM;
    const int wid  = tid >> 5, lane = tid & 31;
    const int wm   = wid >> 1, wn = wid & 1;      // 4x2 wave grid
    const int half = lane >> 4, l15 = lane & 15;

    v8f acc[2][2] = {};

    for (int k0 = 0; k0 < D_; k0 += BK) {
        __syncthreads();
        if (A_BF16) {
            // async global->LDS, 16B per lane, 2 per thread (128x32 bf16)
            const __bf16* A = (const __bf16*)Ap;
            #pragma unroll
            for (int i = 0; i < 2; ++i) {
                int e = tid + i * 256;
                int row = e >> 2, c8 = (e & 3) << 3;
                async_copy_b128(&A[(size_t)(m0 + row) * D_ + k0 + c8],
                                &As[row * LDA + c8]);
            }
        } else {
            const float* A = (const float*)Ap;
            #pragma unroll
            for (int i = 0; i < 4; ++i) {
                int e = tid + i * 256;
                int row = e >> 3, kc = (e & 7) << 2;
                float4 val = *(const float4*)&A[(size_t)(m0 + row) * D_ + k0 + kc];
                __bf16* p = &As[row * LDA + kc];
                p[0] = (__bf16)val.x; p[1] = (__bf16)val.y;
                p[2] = (__bf16)val.z; p[3] = (__bf16)val.w;
            }
        }
        // B tile (fp32 -> bf16), transposed into LDS so B-frags are contiguous
        #pragma unroll
        for (int i = 0; i < 2; ++i) {
            int e = tid + i * 256;
            int kk = e >> 4, nc = (e & 15) << 2;
            float4 val = *(const float4*)&W[(size_t)(k0 + kk) * D_ + n0 + nc];
            Bs[(nc + 0) * LDB + kk] = (__bf16)val.x;
            Bs[(nc + 1) * LDB + kk] = (__bf16)val.y;
            Bs[(nc + 2) * LDB + kk] = (__bf16)val.z;
            Bs[(nc + 3) * LDB + kk] = (__bf16)val.w;
        }
        if (A_BF16) wait_async0();
        __syncthreads();

        v16bf a[2], b[2];
        #pragma unroll
        for (int t = 0; t < 2; ++t) {
            // A frag: row = lane&15; K = half*16 .. +16 contiguous
            const __bf16* pa = &As[(wm * 32 + t * 16 + l15) * LDA + half * 16];
            union { v16bf v; v8bf h2[2]; } u;
            u.h2[0] = *(const v8bf*)pa;
            u.h2[1] = *(const v8bf*)(pa + 8);
            a[t] = u.v;
            // B frag: col = lane&15; K = half*16 .. +16 contiguous
            b[t] = *(const v16bf*)&Bs[(wn * 32 + t * 16 + l15) * LDB + half * 16];
        }
        #pragma unroll
        for (int mt = 0; mt < 2; ++mt)
            #pragma unroll
            for (int nt = 0; nt < 2; ++nt)
                acc[mt][nt] = __builtin_amdgcn_wmma_f32_16x16x32_bf16(
                    false, a[mt], false, b[nt], (short)0, acc[mt][nt], false, false);
    }

    // epilogue: C layout -> row = r + 8*half, col = lane&15
    #pragma unroll
    for (int mt = 0; mt < 2; ++mt) {
        #pragma unroll
        for (int nt = 0; nt < 2; ++nt) {
            int col = n0 + wn * 32 + nt * 16 + l15;
            float bval = bias[col];
            #pragma unroll
            for (int r = 0; r < 8; ++r) {
                int row = m0 + wm * 32 + mt * 16 + r + 8 * half;
                float vv = acc[mt][nt][r] + bval;
                if (SPLIT_HEADS) {
                    __bf16* out = (__bf16*)outp;     // [B,H,S,DK] bf16
                    int bb = row >> 11, ss = row & (S_ - 1);
                    int hh = col >> 6,  dk = col & 63;
                    out[((size_t)(bb * H_ + hh) * S_ + ss) * DK_ + dk] = (__bf16)vv;
                } else {
                    float* out = (float*)outp;       // [M,D] fp32
                    out[(size_t)row * D_ + col] = vv;
                }
            }
        }
    }
}

// ---------------- Flash attention (causal) ----------------
constexpr int LQ = 72, LK = 80, LVT = 48, LP = 40;

__global__ __launch_bounds__(256)
void attention_kernel(const __bf16* __restrict__ Q,
                      const __bf16* __restrict__ K,
                      const __bf16* __restrict__ V,
                      __bf16* __restrict__ ctx)
{
    __shared__ __bf16 Qs[128 * LQ];       // 128 q rows x 64 dk
    __shared__ __bf16 Ks[32 * LK];        // 32 keys x 64 dk (row major)
    __shared__ __bf16 Vt[64 * LVT];       // transposed: Vt[dk][key]
    __shared__ __bf16 Ps[8 * 16 * LP];    // per-wave P tile 16x32

    const int tid  = threadIdx.x;
    const int wid  = tid >> 5, lane = tid & 31;
    const int half = lane >> 4, l15 = lane & 15;
    const int qb   = blockIdx.x;
    const int bh   = blockIdx.y;
    const int b    = bh / H_, h = bh % H_;

    const size_t head_base = (size_t)(b * H_ + h) * S_ * DK_;
    const int q_start = qb * 128;
    const int q0 = q_start + wid * 16;    // this wave's first q row

    // stage Q (contiguous 128*64 bf16) via async copies
    {
        const __bf16* src = Q + head_base + (size_t)q_start * DK_;
        #pragma unroll
        for (int i = 0; i < 4; ++i) {
            int e = tid + i * 256;
            int row = e >> 3, c8 = (e & 7) << 3;
            async_copy_b128(&src[row * DK_ + c8], &Qs[row * LQ + c8]);
        }
    }
    wait_async0();
    __syncthreads();

    v16bf qa[2];                           // Q A-fragments, dk chunks of 32
    #pragma unroll
    for (int c = 0; c < 2; ++c) {
        const __bf16* p = &Qs[(wid * 16 + l15) * LQ + c * 32 + half * 16];
        union { v16bf v; v8bf h2[2]; } u;
        u.h2[0] = *(const v8bf*)p;
        u.h2[1] = *(const v8bf*)(p + 8);
        qa[c] = u.v;
    }

    v8f acc[4] = {};
    float m_i[8], l_i[8];
    #pragma unroll
    for (int r = 0; r < 8; ++r) { m_i[r] = -1e30f; l_i[r] = 0.f; }

    const int numkb = (q_start + 128) / 32;
    const float sc  = 0.125f;             // 1/sqrt(DK)
    const float L2E = 1.44269504f;

    for (int j = 0; j < numkb; ++j) {
        const int kb = j * 32;
        __syncthreads();                  // compute of prev iter done
        {   // stage K tile (32x64, contiguous): async global->LDS
            const __bf16* src = K + head_base + (size_t)kb * DK_;
            int row = tid >> 3, c8 = (tid & 7) << 3;
            async_copy_b128(&src[row * DK_ + c8], &Ks[row * LK + c8]);
        }
        {   // stage V tile transposed (conversion-free but reordered)
            const __bf16* src = V + head_base + (size_t)kb * DK_;
            #pragma unroll
            for (int i = 0; i < 8; ++i) {
                int e = tid + i * 256;
                int r = e >> 6, dk = e & 63;
                Vt[dk * LVT + r] = src[r * DK_ + dk];
            }
        }
        // prefetch next key/value tiles into L2 while we compute
        if (j + 1 < numkb) {
            const __bf16* nk = K + head_base + (size_t)(kb + 32) * DK_;
            const __bf16* nv = V + head_base + (size_t)(kb + 32) * DK_;
            __builtin_prefetch(nk + tid * 16, 0, 1);
            __builtin_prefetch(nv + tid * 16, 0, 1);
        }
        wait_async0();
        __syncthreads();

        if (kb > q0 + 15) continue;       // fully masked; still hits barriers

        // scores: S[16x32] = Q[16x64] @ K_tile^T
        v8f s0 = {}, s1 = {};
        #pragma unroll
        for (int c = 0; c < 2; ++c) {
            v16bf kf0 = *(const v16bf*)&Ks[(l15)      * LK + c * 32 + half * 16];
            v16bf kf1 = *(const v16bf*)&Ks[(16 + l15) * LK + c * 32 + half * 16];
            s0 = __builtin_amdgcn_wmma_f32_16x16x32_bf16(false, qa[c], false, kf0,
                                                         (short)0, s0, false, false);
            s1 = __builtin_amdgcn_wmma_f32_16x16x32_bf16(false, qa[c], false, kf1,
                                                         (short)0, s1, false, false);
        }

        float p0[8], p1[8], alpha[8];
        #pragma unroll
        for (int r = 0; r < 8; ++r) {
            int i = q0 + r + 8 * half;    // global query index for this slot
            float v0 = s0[r] * sc, v1 = s1[r] * sc;
            if (kb + l15      > i) v0 = -1e30f;   // causal mask
            if (kb + 16 + l15 > i) v1 = -1e30f;
            float tmax = fmaxf(v0, v1);
            #pragma unroll
            for (int off = 8; off >= 1; off >>= 1)
                tmax = fmaxf(tmax, __shfl_xor(tmax, off, 32));
            float mnew = fmaxf(m_i[r], tmax);
            alpha[r] = exp2f((m_i[r] - mnew) * L2E);
            p0[r] = exp2f((v0 - mnew) * L2E);
            p1[r] = exp2f((v1 - mnew) * L2E);
            float ts = p0[r] + p1[r];
            #pragma unroll
            for (int off = 8; off >= 1; off >>= 1)
                ts += __shfl_xor(ts, off, 32);
            l_i[r] = l_i[r] * alpha[r] + ts;
            m_i[r] = mnew;
        }
        #pragma unroll
        for (int c = 0; c < 4; ++c)
            #pragma unroll
            for (int r = 0; r < 8; ++r)
                acc[c][r] *= alpha[r];

        // C-layout -> A-layout via per-wave LDS slice (in-order DS, no barrier)
        __bf16* P = &Ps[wid * 16 * LP];
        #pragma unroll
        for (int r = 0; r < 8; ++r) {
            int row = r + 8 * half;
            P[row * LP + l15]      = (__bf16)p0[r];
            P[row * LP + 16 + l15] = (__bf16)p1[r];
        }
        v16bf pa;
        {
            const __bf16* pp = &P[l15 * LP + half * 16];
            union { v16bf v; v8bf h2[2]; } u;
            u.h2[0] = *(const v8bf*)pp;
            u.h2[1] = *(const v8bf*)(pp + 8);
            pa = u.v;
        }
        // O[16x64] += P[16x32] @ V[32x64]
        #pragma unroll
        for (int c = 0; c < 4; ++c) {
            v16bf vf = *(const v16bf*)&Vt[(c * 16 + l15) * LVT + half * 16];
            acc[c] = __builtin_amdgcn_wmma_f32_16x16x32_bf16(false, pa, false, vf,
                                                             (short)0, acc[c], false, false);
        }
    }

    // normalize, write context bf16 in [B*S, D] (heads concatenated)
    #pragma unroll
    for (int c = 0; c < 4; ++c) {
        #pragma unroll
        for (int r = 0; r < 8; ++r) {
            int i = q0 + r + 8 * half;
            float o = acc[c][r] / l_i[r];
            ctx[((size_t)(b * S_ + i)) * D_ + h * DK_ + c * 16 + l15] = (__bf16)o;
        }
    }
}

// ---------------- launch ----------------
extern "C" void kernel_launch(void* const* d_in, const int* in_sizes, int n_in,
                              void* d_out, int out_size, void* d_ws, size_t ws_size,
                              hipStream_t stream)
{
    const float* q  = (const float*)d_in[0];
    const float* k  = (const float*)d_in[1];
    const float* v  = (const float*)d_in[2];
    // d_in[3] = causal mask (int) — applied analytically, not read
    const float* wq = (const float*)d_in[4];
    const float* bq = (const float*)d_in[5];
    const float* wk = (const float*)d_in[6];
    const float* bk = (const float*)d_in[7];
    const float* wv = (const float*)d_in[8];
    const float* bv = (const float*)d_in[9];
    const float* wo = (const float*)d_in[10];
    const float* bo = (const float*)d_in[11];

    char* ws = (char*)d_ws;
    const size_t seg = (size_t)M_ * D_ * sizeof(__bf16);   // 8 MB
    __bf16* Qbf = (__bf16*)(ws);
    __bf16* Kbf = (__bf16*)(ws + seg);
    __bf16* Vbf = (__bf16*)(ws + 2 * seg);
    __bf16* ctx = (__bf16*)(ws + 3 * seg);

    dim3 gg(D_ / BN, M_ / BM);            // (16, 32)
    gemm_bias_kernel<false, true><<<gg, 256, 0, stream>>>(q, wq, bq, Qbf);
    gemm_bias_kernel<false, true><<<gg, 256, 0, stream>>>(k, wk, bk, Kbf);
    gemm_bias_kernel<false, true><<<gg, 256, 0, stream>>>(v, wv, bv, Vbf);

    dim3 ga(S_ / 128, B_ * H_);           // (16, 32)
    attention_kernel<<<ga, 256, 0, stream>>>(Qbf, Kbf, Vbf, ctx);

    gemm_bias_kernel<true, false><<<gg, 256, 0, stream>>>(ctx, wo, bo, d_out);
}
